// SimplifiedHGTEncoder_21732534518025
// MI455X (gfx1250) — compile-verified, matrix operand order checked by reading, not run on hardware
//
#include <hip/hip_runtime.h>

// ---------------------------------------------------------------------------
// SimplifiedHGTEncoder for MI455X (gfx1250, wave32).
// Dense linears via v_wmma_f32_16x16x32_bf16 (K fully unrolled at compile
// time); edge softmax/aggregation via L2-resident f32 atomics (all per-batch
// tensors << 192MB L2, so gathers/scatters never touch HBM at steady state).
// ---------------------------------------------------------------------------

typedef __attribute__((ext_vector_type(16))) __bf16 v16bf;
typedef __attribute__((ext_vector_type(8)))  float  v8f;
typedef __attribute__((ext_vector_type(8)))  int    v8i_;

#define BB 4
#define NN 10000
#define EE 160000
#define FF 64
#define DD 128
#define OO 128
#define LL 2
#define HH 4
#define DH 32

__device__ __forceinline__ void atomicMaxFloat(float* addr, float val) {
  // Mixed-sign-safe float max via integer atomics (init must be -inf).
  if (!(val < 0.0f)) {
    atomicMax((int*)addr, __float_as_int(val));
  } else {
    atomicMin((unsigned int*)addr, __float_as_uint(val));
  }
}

// ---------------------------------------------------------------------------
// C[b, m, n] = sum_k A[b, m, k] * W[k, n] + bias[n]  (+ type_embed[types[b,m], n])
// A: (B, M, K) fp32 row-major, W: (K, 128) fp32 row-major, C: (B, M, 128).
// Block = 256 threads = 8 waves; wave w owns column tile w (Nc = 128 = 8*16).
// One block per 16-row tile. K is a compile-time multiple of 32.
//
// Fragment packing follows the ISA 16-bit A-matrix 16x32 layout: lanes 0-15
// hold K {0..7, 16..23}, lanes 16-31 hold K {8..15, 24..31}, two bf16 per
// VGPR from consecutive K. Each fragment dword is assembled with two integer
// ops from the pair of fp32 values (truncating bf16 conversion).
// ---------------------------------------------------------------------------
template <int K, bool TYPED>
__launch_bounds__(256)
__global__ void wmma_linear(const float* __restrict__ A,
                            const float* __restrict__ W,
                            const float* __restrict__ bias,
                            const float* __restrict__ type_embed,
                            const int*   __restrict__ ntypes,
                            float* __restrict__ C,
                            int M) {
  const int Nc = DD;
  const int tilesM = M >> 4;
  const int bidx = blockIdx.x / tilesM;
  const int tm   = blockIdx.x % tilesM;
  const int wave = threadIdx.x >> 5;
  const int lane = threadIdx.x & 31;
  const int hf   = lane >> 4;       // which 16-lane half
  const int l16  = lane & 15;

  // Base K (even) for each fragment dword (pair of consecutive K values).
  int kbase[8];
#pragma unroll
  for (int p = 0; p < 8; ++p)
    kbase[p] = (p < 4) ? ((hf ? 8 : 0) + 2 * p)
                       : ((hf ? 24 : 16) + 2 * (p - 4));

  const float* Arow = A + ((size_t)bidx * M + (size_t)tm * 16 + l16) * K;
  const int coln = wave * 16 + l16;

  v8f acc = {};
#pragma unroll
  for (int k0 = 0; k0 < K; k0 += 32) {
    v8i_ ai, bi;
#pragma unroll
    for (int p = 0; p < 8; ++p) {
      int kp = k0 + kbase[p];
      unsigned a0 = __float_as_uint(Arow[kp]);
      unsigned a1 = __float_as_uint(Arow[kp + 1]);
      ai[p] = (int)((a0 >> 16) | (a1 & 0xffff0000u));
      unsigned b0 = __float_as_uint(W[(size_t)kp * Nc + coln]);
      unsigned b1 = __float_as_uint(W[((size_t)kp + 1) * Nc + coln]);
      bi[p] = (int)((b0 >> 16) | (b1 & 0xffff0000u));
    }
    v16bf afrag = __builtin_bit_cast(v16bf, ai);
    v16bf bfrag = __builtin_bit_cast(v16bf, bi);
    acc = __builtin_amdgcn_wmma_f32_16x16x32_bf16(
        /*neg_a=*/false, afrag, /*neg_b=*/false, bfrag,
        /*c_mod=*/(short)0, acc, /*reuse_a=*/false, /*reuse_b=*/false);
  }

  // C/D layout: VGPR r -> M = r (lanes 0-15) or M = 8 + r (lanes 16-31), N = l16.
#pragma unroll
  for (int r = 0; r < 8; ++r) {
    int m = tm * 16 + r + 8 * hf;
    float v = acc[r] + bias[coln];
    if (TYPED) {
      int ty = ntypes[(size_t)bidx * M + m];
      v += type_embed[(size_t)ty * Nc + coln];
    }
    C[((size_t)bidx * M + m) * Nc + coln] = v;
  }
}

// e_s / e_d per (b, n, h): 32-wide dot of g slice with a_src / a_dst.
__global__ void attn_scores(const float* __restrict__ g,
                            const float* __restrict__ a_s,
                            const float* __restrict__ a_d,
                            float* __restrict__ es, float* __restrict__ ed) {
  int t = blockIdx.x * blockDim.x + threadIdx.x;    // (b*N + n)*H + h
  if (t >= BB * NN * HH) return;
  int h = t & (HH - 1);
  int node = t >> 2;
  const float* gp = g + (size_t)node * DD + h * DH;
  const float* as = a_s + h * DH;
  const float* ad = a_d + h * DH;
  float s = 0.f, d = 0.f;
#pragma unroll
  for (int i = 0; i < DH; ++i) {
    float gv = gp[i];
    s += gv * as[i];
    d += gv * ad[i];
  }
  es[t] = s;
  ed[t] = d;
}

__global__ void init_layer(float* __restrict__ m, float* __restrict__ denom,
                           float* __restrict__ agg, long long nh, long long nd) {
  long long t = (long long)blockIdx.x * blockDim.x + threadIdx.x;
  if (t < nd) agg[t] = 0.0f;
  if (t < nh) { m[t] = -INFINITY; denom[t] = 0.0f; }
}

__device__ __forceinline__ float leaky02(float x) {
  return x > 0.0f ? x : 0.2f * x;
}

// Pass A: segment max of edge logits into m[dst, h].
__global__ void edge_max(const int* __restrict__ ei,
                         const float* __restrict__ es,
                         const float* __restrict__ ed,
                         float* __restrict__ m) {
  int t = blockIdx.x * blockDim.x + threadIdx.x;    // b*E + e
  if (t >= BB * EE) return;
  int b = t / EE, e = t - b * EE;
  int src = ei[((size_t)b * 2 + 0) * EE + e];
  int dst = ei[((size_t)b * 2 + 1) * EE + e];
  const float* esp = es + ((size_t)b * NN + src) * HH;
  const float* edp = ed + ((size_t)b * NN + dst) * HH;
  float* mp = m + ((size_t)b * NN + dst) * HH;
#pragma unroll
  for (int h = 0; h < HH; ++h)
    atomicMaxFloat(mp + h, leaky02(esp[h] + edp[h]));
}

// Pass B: ex = exp(e - finite(m[dst])); store ex, accumulate denom[dst].
__global__ void edge_exp(const int* __restrict__ ei,
                         const float* __restrict__ es,
                         const float* __restrict__ ed,
                         const float* __restrict__ m,
                         float* __restrict__ denom,
                         float* __restrict__ exbuf) {
  int t = blockIdx.x * blockDim.x + threadIdx.x;    // b*E + e
  if (t >= BB * EE) return;
  int b = t / EE, e = t - b * EE;
  int src = ei[((size_t)b * 2 + 0) * EE + e];
  int dst = ei[((size_t)b * 2 + 1) * EE + e];
  const float* esp = es + ((size_t)b * NN + src) * HH;
  const float* edp = ed + ((size_t)b * NN + dst) * HH;
  const float* mp  = m  + ((size_t)b * NN + dst) * HH;
  float* dp = denom + ((size_t)b * NN + dst) * HH;
  float* xp = exbuf + (size_t)t * HH;
#pragma unroll
  for (int h = 0; h < HH; ++h) {
    float v = leaky02(esp[h] + edp[h]);
    float mm = mp[h];
    if ((__float_as_uint(mm) & 0x7f800000u) == 0x7f800000u) mm = 0.0f;  // !finite -> 0
    float ex = __expf(v - mm);
    xp[h] = ex;
    atomicAdd(dp + h, ex);
  }
}

// Pass C: agg[dst, h*32+d] += (ex/denom) * g[src, h*32+d]. One thread per (b,e,h).
__global__ void edge_agg(const int* __restrict__ ei,
                         const float* __restrict__ exbuf,
                         const float* __restrict__ denom,
                         const float* __restrict__ g,
                         float* __restrict__ agg) {
  int t = blockIdx.x * blockDim.x + threadIdx.x;    // (b*E + e)*H + h
  if (t >= BB * EE * HH) return;
  int h = t & (HH - 1);
  int be = t >> 2;
  int b = be / EE, e = be - b * EE;
  int src = ei[((size_t)b * 2 + 0) * EE + e];
  int dst = ei[((size_t)b * 2 + 1) * EE + e];
  float dn = denom[((size_t)b * NN + dst) * HH + h];
  float attn = exbuf[(size_t)t] / (dn + 1e-16f);
  const float4* gs = (const float4*)(g + ((size_t)b * NN + src) * DD + h * DH);
  float* ag = agg + ((size_t)b * NN + dst) * DD + h * DH;
#pragma unroll
  for (int q = 0; q < DH / 4; ++q) {
    float4 gv = gs[q];
    atomicAdd(ag + 4 * q + 0, attn * gv.x);
    atomicAdd(ag + 4 * q + 1, attn * gv.y);
    atomicAdd(ag + 4 * q + 2, attn * gv.z);
    atomicAdd(ag + 4 * q + 3, attn * gv.w);
  }
}

// h = elu(agg + h) * mask   (alpha = 1)
__global__ void node_update(float* __restrict__ hbuf,
                            const float* __restrict__ agg,
                            const float* __restrict__ mask) {
  long long t = (long long)blockIdx.x * blockDim.x + threadIdx.x;
  if (t >= (long long)BB * NN * DD) return;
  long long node = t >> 7;   // /128
  float v = agg[t] + hbuf[t];
  v = v > 0.0f ? v : (__expf(v) - 1.0f);
  hbuf[t] = v * mask[node];
}

// graph_emb[b, o] = sum_n node_emb[b,n,o]*mask[b,n] / max(sum_n mask, 1)
__global__ void graph_pool(const float* __restrict__ ne,
                           const float* __restrict__ mask,
                           float* __restrict__ gout) {
  int b = blockIdx.x >> 7;
  int o = blockIdx.x & (OO - 1);
  __shared__ float sb[256];
  __shared__ float mb[256];
  float s = 0.f, ms = 0.f;
  for (int n = threadIdx.x; n < NN; n += 256) {
    float mk = mask[(size_t)b * NN + n];
    s  += ne[((size_t)b * NN + n) * OO + o] * mk;
    ms += mk;
  }
  sb[threadIdx.x] = s;
  mb[threadIdx.x] = ms;
  __syncthreads();
  for (int st = 128; st > 0; st >>= 1) {
    if (threadIdx.x < st) {
      sb[threadIdx.x] += sb[threadIdx.x + st];
      mb[threadIdx.x] += mb[threadIdx.x + st];
    }
    __syncthreads();
  }
  if (threadIdx.x == 0)
    gout[(size_t)b * OO + o] = sb[0] / fmaxf(mb[0], 1.0f);
}

extern "C" void kernel_launch(void* const* d_in, const int* in_sizes, int n_in,
                              void* d_out, int out_size, void* d_ws, size_t ws_size,
                              hipStream_t stream) {
  (void)in_sizes; (void)n_in; (void)out_size; (void)ws_size;

  const float* node_features = (const float*)d_in[0];
  const int*   edge_index    = (const int*)  d_in[1];
  const int*   node_types    = (const int*)  d_in[2];
  const float* node_mask     = (const float*)d_in[3];
  const float* type_embed    = (const float*)d_in[4];
  const float* in_W          = (const float*)d_in[5];
  const float* in_b          = (const float*)d_in[6];
  const float* gat_W         = (const float*)d_in[7];
  const float* gat_b         = (const float*)d_in[8];
  const float* a_src         = (const float*)d_in[9];
  const float* a_dst         = (const float*)d_in[10];
  const float* out_W         = (const float*)d_in[11];
  const float* out_b         = (const float*)d_in[12];
  float* out = (float*)d_out;

  // Workspace carve-up (all fp32). ~74 MB total.
  float* ws = (float*)d_ws;
  size_t off = 0;
  float* hbuf  = ws + off; off += (size_t)BB * NN * DD;   // 5.12M
  float* gbuf  = ws + off; off += (size_t)BB * NN * DD;
  float* aggb  = ws + off; off += (size_t)BB * NN * DD;
  float* esb   = ws + off; off += (size_t)BB * NN * HH;
  float* edb   = ws + off; off += (size_t)BB * NN * HH;
  float* mbuf  = ws + off; off += (size_t)BB * NN * HH;
  float* dbuf  = ws + off; off += (size_t)BB * NN * HH;
  float* exb   = ws + off; off += (size_t)BB * EE * HH;   // 2.56M

  const int tilesM = NN / 16;                      // 625
  dim3 mmGrid(BB * tilesM), mmBlk(256);

  // h = X @ in_W + in_b + type_embed[node_types]   (K = 64, typed epilogue)
  wmma_linear<FF, true><<<mmGrid, mmBlk, 0, stream>>>(
      node_features, in_W, in_b, type_embed, node_types, hbuf, NN);

  const long long nd = (long long)BB * NN * DD;
  const long long nh = (long long)BB * NN * HH;
  const int edgeBlocks  = (BB * EE + 255) / 256;
  const int edgeHBlocks = (BB * EE * HH + 255) / 256;

  for (int l = 0; l < LL; ++l) {
    const float* Wl  = gat_W + (size_t)l * DD * DD;
    const float* bl  = gat_b + (size_t)l * DD;
    const float* asl = a_src + (size_t)l * HH * DH;
    const float* adl = a_dst + (size_t)l * HH * DH;

    // g = h @ W + b   (K = 128)
    wmma_linear<DD, false><<<mmGrid, mmBlk, 0, stream>>>(
        hbuf, Wl, bl, nullptr, nullptr, gbuf, NN);
    attn_scores<<<(BB * NN * HH + 255) / 256, 256, 0, stream>>>(gbuf, asl, adl,
                                                                esb, edb);
    init_layer<<<(int)((nd + 255) / 256), 256, 0, stream>>>(mbuf, dbuf, aggb, nh, nd);
    edge_max<<<edgeBlocks, 256, 0, stream>>>(edge_index, esb, edb, mbuf);
    edge_exp<<<edgeBlocks, 256, 0, stream>>>(edge_index, esb, edb, mbuf, dbuf, exb);
    edge_agg<<<edgeHBlocks, 256, 0, stream>>>(edge_index, exb, dbuf, gbuf, aggb);
    node_update<<<(int)((nd + 255) / 256), 256, 0, stream>>>(hbuf, aggb, node_mask);
  }

  // node_emb = h @ out_W + out_b   (written straight into d_out, K = 128)
  wmma_linear<DD, false><<<mmGrid, mmBlk, 0, stream>>>(
      hbuf, out_W, out_b, nullptr, nullptr, out, NN);
  // graph_emb (after node_emb, same stream)
  graph_pool<<<BB * OO, 256, 0, stream>>>(out, node_mask,
                                          out + (size_t)BB * NN * OO);
}
